// RoPE_mha_61916248539458
// MI455X (gfx1250) — matrix operand order repaired and checked by reading, not compiled, not run on hardware
//
#include <hip/hip_runtime.h>
#include <hip/hip_bf16.h>

typedef __bf16 bf16_t;
typedef __attribute__((ext_vector_type(16))) __bf16 v16bf;
typedef __attribute__((ext_vector_type(8)))  __bf16 bf16x8;
typedef __attribute__((ext_vector_type(8)))  float  v8f;

#define WMMA_BF16(a, b, c) \
  __builtin_amdgcn_wmma_f32_16x16x32_bf16(false, (a), false, (b), (short)0, (c), false, false)

// ---------------------------------------------------------------------------
// Fragment loaders (wave32 WMMA 16x16x32 bf16 layouts as used here; contiguous K order, same in A and B)
// A (16x32, MxK): lane m = lane%16; elems 0-15 -> K = 16*half + 0..15
// B (32x16, KxN): lane n = lane%16; elems 0-15 -> K = 16*half + 0..15
// C/D (16x16 f32): vgpr j -> row j + 8*half, col = lane%16
// ---------------------------------------------------------------------------
__device__ __forceinline__ v16bf load_frag_A(const bf16_t* base, int stride,
                                             int row, int half, int koff) {
  const bf16_t* p = base + (size_t)row * stride + koff;
  bf16x8 lo = *(const bf16x8*)(p + half * 16);
  bf16x8 hi = *(const bf16x8*)(p + half * 16 + 8);
  v16bf r;
  #pragma unroll
  for (int i = 0; i < 8; ++i) { r[i] = lo[i]; r[i + 8] = hi[i]; }
  return r;
}

__device__ __forceinline__ v16bf load_frag_B(const bf16_t* base, int stride,
                                             int row, int half, int koff) {
  const bf16_t* p = base + (size_t)row * stride + koff + 16 * half;
  bf16x8 lo = *(const bf16x8*)(p);
  bf16x8 hi = *(const bf16x8*)(p + 8);
  v16bf r;
  #pragma unroll
  for (int i = 0; i < 8; ++i) { r[i] = lo[i]; r[i + 8] = hi[i]; }
  return r;
}

// Raw LDS byte address = low 32 bits of the generic (flat) pointer to shared.
__device__ __forceinline__ unsigned lds_addr_of(const void* p) {
  return (unsigned)(unsigned long long)p;
}

// CDNA5 async memory->LDS copy (16B per lane), tracked by ASYNCcnt.
__device__ __forceinline__ void async_b128(unsigned lds_byte_addr,
                                           const void* sbase, unsigned voff) {
  asm volatile("global_load_async_to_lds_b128 %0, %1, %2"
               :: "v"(lds_byte_addr), "v"(voff), "s"(sbase)
               : "memory");
}

__device__ __forceinline__ void wait_async0() {
  asm volatile("s_wait_asynccnt 0x0" ::: "memory");
}

// ---------------------------------------------------------------------------
// fp32 -> bf16 conversion (grid-stride)
// ---------------------------------------------------------------------------
__global__ void cvt_kernel(const float* __restrict__ s, bf16_t* __restrict__ d, int n) {
  int i = blockIdx.x * blockDim.x + threadIdx.x;
  int stride = gridDim.x * blockDim.x;
  for (; i < n; i += stride) d[i] = (bf16_t)s[i];
}

// ---------------------------------------------------------------------------
// 128x128 GEMM main loop: C = X[M x 1024] * W[N x 1024]^T
// 256 threads = 8 waves (2 M x 4 N), each wave does 4x2 16x16 WMMA tiles.
// Double-buffered LDS tiles filled by GLOBAL_LOAD_ASYNC_TO_LDS_B128.
// As/Bs each hold 2 buffers of 128x32 bf16 (8 KB), buffer stride 4096 elems.
// ---------------------------------------------------------------------------
__device__ __forceinline__ void gemm_tile_128x128(
    const bf16_t* __restrict__ X, const bf16_t* __restrict__ W,
    int m0, int n0, bf16_t* As, bf16_t* Bs, v8f acc[4][2]) {
  int tid  = threadIdx.x;
  int wave = tid >> 5, lane = tid & 31;
  int half = lane >> 4, ln = lane & 15;
  int waveM = wave >> 2, waveN = wave & 3;

  #pragma unroll
  for (int i = 0; i < 4; ++i)
    #pragma unroll
    for (int j = 0; j < 2; ++j)
      #pragma unroll
      for (int r = 0; r < 8; ++r) acc[i][j][r] = 0.0f;

  int arow = tid >> 1;
  int acol = (tid & 1) * 16;
  unsigned lds_a = lds_addr_of(As) + (unsigned)(arow * 32 + acol) * 2u;
  unsigned lds_b = lds_addr_of(Bs) + (unsigned)(arow * 32 + acol) * 2u;
  unsigned xrow = (unsigned)((m0 + arow) * 1024 + acol) * 2u;  // byte offset
  unsigned wrow = (unsigned)((n0 + arow) * 1024 + acol) * 2u;  // byte offset

#define ISSUE_TILE(kc_, p_)                                          \
  do {                                                               \
    unsigned ka = xrow + (unsigned)(kc_) * 2u;                       \
    unsigned kb = wrow + (unsigned)(kc_) * 2u;                       \
    async_b128(lds_a + (unsigned)(p_) * 8192u,       X, ka);         \
    async_b128(lds_a + (unsigned)(p_) * 8192u + 16u, X, ka + 16u);   \
    async_b128(lds_b + (unsigned)(p_) * 8192u,       W, kb);         \
    async_b128(lds_b + (unsigned)(p_) * 8192u + 16u, W, kb + 16u);   \
  } while (0)

  ISSUE_TILE(0, 0);
  int p = 0;
  for (int kc = 0; kc < 1024; kc += 32) {
    wait_async0();      // this wave's tile-kc async writes have landed
    __syncthreads();    // ... and every other wave's too
    if (kc + 32 < 1024) ISSUE_TILE(kc + 32, p ^ 1);

    const bf16_t* Ap = As + p * 4096;
    const bf16_t* Bp = Bs + p * 4096;
    v16bf af[4], bfr[2];
    #pragma unroll
    for (int i = 0; i < 4; ++i)
      af[i] = load_frag_A(Ap, 32, waveM * 64 + i * 16 + ln, half, 0);
    #pragma unroll
    for (int j = 0; j < 2; ++j)
      bfr[j] = load_frag_B(Bp, 32, waveN * 32 + j * 16 + ln, half, 0);

    #pragma unroll
    for (int i = 0; i < 4; ++i)
      #pragma unroll
      for (int j = 0; j < 2; ++j)
        acc[i][j] = WMMA_BF16(af[i], bfr[j], acc[i][j]);
    p ^= 1;
  }
#undef ISSUE_TILE
}

// QKV projection: writes bf16 into [B,H,S,Dh] layout, +bias.
__global__ void gemm_qkv_kernel(const bf16_t* __restrict__ X,
                                const bf16_t* __restrict__ W,
                                const float* __restrict__ bias,
                                bf16_t* __restrict__ dst) {
  __shared__ __align__(16) bf16_t As[2 * 128 * 32];
  __shared__ __align__(16) bf16_t Bs[2 * 128 * 32];
  v8f acc[4][2];
  int m0 = blockIdx.y * 128, n0 = blockIdx.x * 128;
  gemm_tile_128x128(X, W, m0, n0, As, Bs, acc);

  int tid = threadIdx.x, wave = tid >> 5, lane = tid & 31;
  int half = lane >> 4, ln = lane & 15;
  int waveM = wave >> 2, waveN = wave & 3;
  #pragma unroll
  for (int i = 0; i < 4; ++i)
    #pragma unroll
    for (int j = 0; j < 2; ++j) {
      int col = n0 + waveN * 32 + j * 16 + ln;
      float b = bias[col];
      int h = col >> 6, dh = col & 63;
      #pragma unroll
      for (int r = 0; r < 8; ++r) {
        int row = m0 + waveM * 64 + i * 16 + r + 8 * half;  // row = b*S + s
        int bb = row >> 11, s = row & 2047;
        dst[(((size_t)(bb * 16 + h) * 2048 + s) << 6) + dh] = (bf16_t)(acc[i][j][r] + b);
      }
    }
}

// Output projection: writes fp32 row-major + bias.
__global__ void gemm_out_kernel(const bf16_t* __restrict__ X,
                                const bf16_t* __restrict__ W,
                                const float* __restrict__ bias,
                                float* __restrict__ out) {
  __shared__ __align__(16) bf16_t As[2 * 128 * 32];
  __shared__ __align__(16) bf16_t Bs[2 * 128 * 32];
  v8f acc[4][2];
  int m0 = blockIdx.y * 128, n0 = blockIdx.x * 128;
  gemm_tile_128x128(X, W, m0, n0, As, Bs, acc);

  int tid = threadIdx.x, wave = tid >> 5, lane = tid & 31;
  int half = lane >> 4, ln = lane & 15;
  int waveM = wave >> 2, waveN = wave & 3;
  #pragma unroll
  for (int i = 0; i < 4; ++i)
    #pragma unroll
    for (int j = 0; j < 2; ++j) {
      int col = n0 + waveN * 32 + j * 16 + ln;
      float b = bias[col];
      #pragma unroll
      for (int r = 0; r < 8; ++r) {
        int row = m0 + waveM * 64 + i * 16 + r + 8 * half;
        out[(size_t)row * 1024 + col] = acc[i][j][r] + b;
      }
    }
}

// ---------------------------------------------------------------------------
// RoPE (half-split): operates in-place on bf16 [B,H,S,64], one thread per pair.
// ---------------------------------------------------------------------------
__global__ void rope_kernel(bf16_t* __restrict__ q) {
  long idx = (long)blockIdx.x * blockDim.x + threadIdx.x;  // B*H*S*32 threads
  int  i = (int)(idx & 31);
  long t = idx >> 5;
  int  s = (int)(t & 2047);
  long bh = t >> 11;
  // inv_freq = 10000^(-2i/64) = exp(-i * ln(10000)/32)
  float ang = (float)s * __expf(-(float)i * 0.2878231366242557f);
  float c, sn;
  __sincosf(ang, &sn, &c);
  size_t base = (size_t)((bh << 11) + s) << 6;
  float x1 = (float)q[base + i];
  float x2 = (float)q[base + i + 32];
  q[base + i]      = (bf16_t)(x1 * c - x2 * sn);
  q[base + i + 32] = (bf16_t)(x2 * c + x1 * sn);
}

// ---------------------------------------------------------------------------
// Causal flash attention: block=128 (4 waves), wave handles one 16-row q-tile.
// Q,K,V bf16 [B,H,S,64]; O bf16 [B,S,H*64] (ready for output projection).
// ---------------------------------------------------------------------------
__global__ void attn_kernel(const bf16_t* __restrict__ Q,
                            const bf16_t* __restrict__ K,
                            const bf16_t* __restrict__ V,
                            bf16_t* __restrict__ O) {
  __shared__ __align__(16) bf16_t Pbuf[4][16 * 32];
  __shared__ __align__(16) bf16_t Vbuf[4][32 * 64];

  int tid = threadIdx.x, wave = tid >> 5, lane = tid & 31;
  int half = lane >> 4, ln = lane & 15;
  int bh = blockIdx.y;
  int q0 = blockIdx.x * 64 + wave * 16;

  const bf16_t* Qh = Q + (size_t)bh * 2048 * 64;
  const bf16_t* Kh = K + (size_t)bh * 2048 * 64;
  const bf16_t* Vh = V + (size_t)bh * 2048 * 64;

  v16bf aq[2];
  #pragma unroll
  for (int c = 0; c < 2; ++c)
    aq[c] = load_frag_A(Qh, 64, q0 + ln, half, c * 32);

  v8f o[4];
  #pragma unroll
  for (int d = 0; d < 4; ++d)
    #pragma unroll
    for (int r = 0; r < 8; ++r) o[d][r] = 0.0f;
  float m_[8], l_[8];
  #pragma unroll
  for (int r = 0; r < 8; ++r) { m_[r] = -1e30f; l_[r] = 0.0f; }

  int kmax = q0 + 15;
  for (int k0 = 0; k0 <= kmax; k0 += 32) {
    // ---- S = Q K^T (scaled) with causal mask ----
    float sv[2][8];
    #pragma unroll
    for (int jn = 0; jn < 2; ++jn) {
      v8f sa;
      #pragma unroll
      for (int r = 0; r < 8; ++r) sa[r] = 0.0f;
      #pragma unroll
      for (int c = 0; c < 2; ++c) {
        v16bf bk = load_frag_B(Kh, 64, k0 + jn * 16 + ln, half, c * 32);
        sa = WMMA_BF16(aq[c], bk, sa);
      }
      int kc = k0 + jn * 16 + ln;
      #pragma unroll
      for (int r = 0; r < 8; ++r) {
        int qr = q0 + r + 8 * half;
        float v = sa[r] * 0.125f;  // 1/sqrt(64)
        sv[jn][r] = (kc > qr) ? -1e30f : v;
      }
    }

    // ---- stage V tile (32 keys x 64) into per-wave LDS ----
    {
      const bf16_t* vrow = Vh + (size_t)(k0 + lane) * 64;
      bf16_t* dv = &Vbuf[wave][lane * 64];
      #pragma unroll
      for (int c = 0; c < 8; ++c)
        *(bf16x8*)(dv + c * 8) = *(const bf16x8*)(vrow + c * 8);
    }

    // ---- online softmax update ----
    float alpha[8];
    #pragma unroll
    for (int r = 0; r < 8; ++r) {
      float t = fmaxf(sv[0][r], sv[1][r]);
      #pragma unroll
      for (int mm = 1; mm < 16; mm <<= 1) t = fmaxf(t, __shfl_xor(t, mm, 32));
      float mn = fmaxf(m_[r], t);
      float p0 = __expf(sv[0][r] - mn);
      float p1 = __expf(sv[1][r] - mn);
      float ps = p0 + p1;
      #pragma unroll
      for (int mm = 1; mm < 16; mm <<= 1) ps += __shfl_xor(ps, mm, 32);
      alpha[r] = __expf(m_[r] - mn);
      l_[r] = l_[r] * alpha[r] + ps;
      m_[r] = mn;
      sv[0][r] = p0;
      sv[1][r] = p1;
    }

    // ---- write P (bf16) to LDS for C-layout -> A-layout relayout ----
    #pragma unroll
    for (int jn = 0; jn < 2; ++jn)
      #pragma unroll
      for (int r = 0; r < 8; ++r)
        Pbuf[wave][(r + 8 * half) * 32 + jn * 16 + ln] = (bf16_t)sv[jn][r];

    // rescale O accumulators
    #pragma unroll
    for (int d = 0; d < 4; ++d)
      #pragma unroll
      for (int r = 0; r < 8; ++r) o[d][r] *= alpha[r];

    asm volatile("s_wait_dscnt 0x0" ::: "memory");  // same-wave LDS RAW

    // ---- O += P @ V ----
    v16bf ap = load_frag_A(&Pbuf[wave][0], 32, ln, half, 0);
    #pragma unroll
    for (int d = 0; d < 4; ++d) {
      v16bf bv;
      #pragma unroll
      for (int e = 0; e < 16; ++e)
        bv[e] = Vbuf[wave][(16 * half + e) * 64 + d * 16 + ln];
      o[d] = WMMA_BF16(ap, bv, o[d]);
    }
  }

  // ---- normalize and write O in [B,S,H*64] bf16 ----
  int b = bh >> 4, h = bh & 15;
  #pragma unroll
  for (int r = 0; r < 8; ++r) {
    float inv = 1.0f / l_[r];
    int s = q0 + r + 8 * half;
    #pragma unroll
    for (int d = 0; d < 4; ++d)
      O[(size_t)(b * 2048 + s) * 1024 + h * 64 + d * 16 + ln] = (bf16_t)(o[d][r] * inv);
  }
}

// ---------------------------------------------------------------------------
extern "C" void kernel_launch(void* const* d_in, const int* in_sizes, int n_in,
                              void* d_out, int out_size, void* d_ws, size_t ws_size,
                              hipStream_t stream) {
  const float* x  = (const float*)d_in[0];
  const float* Wq = (const float*)d_in[1];
  const float* bq = (const float*)d_in[2];
  const float* Wk = (const float*)d_in[3];
  const float* bk = (const float*)d_in[4];
  const float* Wv = (const float*)d_in[5];
  const float* bv = (const float*)d_in[6];
  const float* Wo = (const float*)d_in[7];
  const float* bo = (const float*)d_in[8];
  // d_in[9] = mask (causal, implemented analytically)

  const size_t NX = (size_t)8192 * 1024;   // B*S x D
  const size_t NW = (size_t)1024 * 1024;   // D x D
  const size_t NQ = (size_t)4 * 16 * 2048 * 64;

  bf16_t* ws  = (bf16_t*)d_ws;
  size_t off  = 0;
  bf16_t* Xbf = ws + off; off += NX;
  bf16_t* Wqb = ws + off; off += NW;
  bf16_t* Wkb = ws + off; off += NW;
  bf16_t* Wvb = ws + off; off += NW;
  bf16_t* Wob = ws + off; off += NW;
  bf16_t* Qb  = ws + off; off += NQ;
  bf16_t* Kb  = ws + off; off += NQ;
  bf16_t* Vb  = ws + off; off += NQ;
  bf16_t* Ob  = ws + off; off += NX;

  cvt_kernel<<<4096, 256, 0, stream>>>(x,  Xbf, (int)NX);
  cvt_kernel<<<1024, 256, 0, stream>>>(Wq, Wqb, (int)NW);
  cvt_kernel<<<1024, 256, 0, stream>>>(Wk, Wkb, (int)NW);
  cvt_kernel<<<1024, 256, 0, stream>>>(Wv, Wvb, (int)NW);
  cvt_kernel<<<1024, 256, 0, stream>>>(Wo, Wob, (int)NW);

  dim3 gg(8, 64);  // N/128, M/128
  gemm_qkv_kernel<<<gg, 256, 0, stream>>>(Xbf, Wqb, bq, Qb);
  gemm_qkv_kernel<<<gg, 256, 0, stream>>>(Xbf, Wkb, bk, Kb);
  gemm_qkv_kernel<<<gg, 256, 0, stream>>>(Xbf, Wvb, bv, Vb);

  rope_kernel<<<16384, 256, 0, stream>>>(Qb);  // B*H*S*32 threads
  rope_kernel<<<16384, 256, 0, stream>>>(Kb);

  attn_kernel<<<dim3(32, 64), 128, 0, stream>>>(Qb, Kb, Vb, Ob);

  gemm_out_kernel<<<gg, 256, 0, stream>>>(Ob, Wob, bo, (float*)d_out);
}
